// DynamicEpidermis_39608188403878
// MI455X (gfx1250) — matrix operand order repaired and checked by reading, not compiled, not run on hardware
//
#include <hip/hip_runtime.h>
#include <math.h>

// Problem dims
#define B_   1024
#define I_   2048
#define O_   2048
#define C_   64
#define R_   16
#define RC   1024   // R*C
#define K2   1040   // RC + R (wa/wb with bias columns folded in)
#define K2P  1056   // K2 padded to multiple of 32
#define NTP  1152   // T columns padded to multiple of 128

typedef __attribute__((ext_vector_type(16))) __bf16 v16bf;
typedef __attribute__((ext_vector_type(8)))  __bf16 bf16x8;
typedef __attribute__((ext_vector_type(8)))  float  v8f;

#if __has_builtin(__builtin_amdgcn_global_load_async_to_lds_b128)
#define USE_ASYNC_LDS 1
#else
#define USE_ASYNC_LDS 0
#endif

#if USE_ASYNC_LDS
// Builtin signature (from hipcc diagnostic): param0 = int vec4 in AS1 (__device__),
// param1 = LDS side, then imm offset + cpol.
typedef int i32v4 __attribute__((vector_size(16)));
typedef __attribute__((address_space(1))) i32v4* as1_i32v4_t;
typedef __attribute__((address_space(3))) i32v4* as3_i32v4_t;
#define ASYNC_G2LDS_B128(gp, lp) \
    __builtin_amdgcn_global_load_async_to_lds_b128((as1_i32v4_t)(gp), (as3_i32v4_t)(lp), 0, 0)
#endif

// float -> bf16 (round-to-nearest-even), stored as ushort to keep scalar math in u32
__device__ __forceinline__ unsigned short f2bf(float f) {
    union { float f; unsigned u; } v; v.f = f;
    unsigned u = v.u;
    unsigned r = u + 0x7FFFu + ((u >> 16) & 1u);
    return (unsigned short)(r >> 16);
}

// ---------------------------------------------------------------------------
// Kernel 1: per-row standardize (ddof=1) + emit x and xn as bf16
// ---------------------------------------------------------------------------
__global__ __launch_bounds__(256) void k_standardize(
        const float* __restrict__ x,
        unsigned short* __restrict__ xb,
        unsigned short* __restrict__ xnb) {
    __shared__ float s0[256], s1[256];
    const int b = blockIdx.x, t = threadIdx.x;
    const float* row = x + (long)b * I_;
    float sum = 0.f, sq = 0.f;
    for (int i = t; i < I_; i += 256) { float v = row[i]; sum += v; sq += v * v; }
    s0[t] = sum; s1[t] = sq; __syncthreads();
    for (int ofs = 128; ofs > 0; ofs >>= 1) {
        if (t < ofs) { s0[t] += s0[t + ofs]; s1[t] += s1[t + ofs]; }
        __syncthreads();
    }
    const float mean = s0[0] / (float)I_;
    float var = (s1[0] - (float)I_ * mean * mean) / (float)(I_ - 1);
    var = fmaxf(var, 0.f);
    const float inv = 1.f / (sqrtf(var) + 1e-8f);
    for (int i = t; i < I_; i += 256) {
        float v = row[i];
        xb [(long)b * I_ + i] = f2bf(v);
        xnb[(long)b * I_ + i] = f2bf((v - mean) * inv);
    }
}

// ---------------------------------------------------------------------------
// Elementwise f32 -> bf16 conversion (grid-stride)
// ---------------------------------------------------------------------------
__global__ __launch_bounds__(256) void k_f32_to_bf16(
        const float* __restrict__ in, unsigned short* __restrict__ out, long n) {
    long stride = (long)gridDim.x * blockDim.x;
    for (long i = (long)blockIdx.x * blockDim.x + threadIdx.x; i < n; i += stride)
        out[i] = f2bf(in[i]);
}

// wa (I*R, C) -> wa_t (NTP, I) bf16: row n = r*C+c holds wa[(k*R+r)*C+c] over k;
// rows [RC, K2) hold ba[k*R + (n-RC)]; rows >= K2 are zero padding.
__global__ __launch_bounds__(256) void k_build_wa_t(
        const float* __restrict__ wa, const float* __restrict__ ba,
        unsigned short* __restrict__ out) {
    const long total = (long)NTP * I_;
    const long stride = (long)gridDim.x * blockDim.x;
    for (long idx = (long)blockIdx.x * blockDim.x + threadIdx.x; idx < total; idx += stride) {
        long n = idx / I_, k = idx - n * I_;
        float v = 0.f;
        if (n < RC) {
            long r = n >> 6, c = n & 63;
            v = wa[(k * R_ + r) * C_ + c];
        } else if (n < K2) {
            v = ba[k * R_ + (n - RC)];
        }
        out[idx] = f2bf(v);
    }
}

// wb (O*R, C) viewed as (O, RC) is contiguous; append bb as cols [RC,K2); pad to K2P.
__global__ __launch_bounds__(256) void k_build_wb_aug(
        const float* __restrict__ wb, const float* __restrict__ bb,
        unsigned short* __restrict__ out) {
    const long total = (long)O_ * K2P;
    const long stride = (long)gridDim.x * blockDim.x;
    for (long idx = (long)blockIdx.x * blockDim.x + threadIdx.x; idx < total; idx += stride) {
        long o = idx / K2P, k = idx - o * K2P;
        float v = 0.f;
        if (k < RC)       v = wb[o * RC + k];
        else if (k < K2)  v = bb[o * R_ + (k - RC)];
        out[idx] = f2bf(v);
    }
}

// ---------------------------------------------------------------------------
// Generic bf16 WMMA GEMM: out[M,N] (+)= act(A[M,K] @ W[N,K]^T + bias)
// 256 threads = 8 waves; block tile 128x128; wave tile 64x32; K step 32.
// Double-buffered LDS; stage n+1 streams in (async DMA to LDS when available)
// while WMMA consumes stage n. One barrier per K-step.
// All dims assumed multiples of the tile (buffers pre-padded) -> EXEC all-1s.
// ---------------------------------------------------------------------------
template<bool ACCUM, bool RELU, bool HAS_BIAS>
__global__ __launch_bounds__(256) void k_gemm_bf16(
        const unsigned short* __restrict__ A,
        const unsigned short* __restrict__ W,
        const float* __restrict__ bias,
        float* __restrict__ out, int N, int K) {
    constexpr int LDT = 40;  // 32 cols + pad (80B row stride, 16B aligned)
    __shared__ __align__(16) unsigned short ldsA[2][128 * LDT];
    __shared__ __align__(16) unsigned short ldsW[2][128 * LDT];

    const int tid  = threadIdx.x;
    const int lane = tid & 31;
    const int wid  = tid >> 5;
    const int half = lane >> 4;    // 0..1
    const int lrow = lane & 15;    // 0..15
    const int wm = (wid & 1) * 64; // wave M offset in block tile
    const int wn = (wid >> 1) * 32;// wave N offset in block tile
    const long blockM = (long)blockIdx.y * 128;
    const long blockN = (long)blockIdx.x * 128;

    v8f acc[4][2];
    #pragma unroll
    for (int i = 0; i < 4; i++)
        #pragma unroll
        for (int j = 0; j < 2; j++)
            #pragma unroll
            for (int e = 0; e < 8; e++) acc[i][j][e] = 0.f;

    // Each 128x32 bf16 tile = 512 chunks of 16B; 256 threads -> 2 chunks/thread.
    const int c0 = tid, c1 = tid + 256;
    const int r0 = c0 >> 2, q0 = (c0 & 3) * 8;
    const int r1 = c1 >> 2, q1 = (c1 & 3) * 8;

    auto stage = [&](int buf, int k0) {
#if USE_ASYNC_LDS
        ASYNC_G2LDS_B128(&A[(blockM + r0) * (long)K + k0 + q0], &ldsA[buf][r0 * LDT + q0]);
        ASYNC_G2LDS_B128(&A[(blockM + r1) * (long)K + k0 + q1], &ldsA[buf][r1 * LDT + q1]);
        ASYNC_G2LDS_B128(&W[(blockN + r0) * (long)K + k0 + q0], &ldsW[buf][r0 * LDT + q0]);
        ASYNC_G2LDS_B128(&W[(blockN + r1) * (long)K + k0 + q1], &ldsW[buf][r1 * LDT + q1]);
#else
        *(uint4*)&ldsA[buf][r0 * LDT + q0] = *(const uint4*)&A[(blockM + r0) * (long)K + k0 + q0];
        *(uint4*)&ldsA[buf][r1 * LDT + q1] = *(const uint4*)&A[(blockM + r1) * (long)K + k0 + q1];
        *(uint4*)&ldsW[buf][r0 * LDT + q0] = *(const uint4*)&W[(blockN + r0) * (long)K + k0 + q0];
        *(uint4*)&ldsW[buf][r1 * LDT + q1] = *(const uint4*)&W[(blockN + r1) * (long)K + k0 + q1];
#endif
    };

    const int nk = K >> 5;  // K / 32
    stage(0, 0);

    for (int n = 0; n < nk; ++n) {
        const int buf = n & 1;
#if USE_ASYNC_LDS
        // My async writes for stage n are complete; barrier makes them (and
        // everyone's) visible, and guarantees all waves finished reading the
        // buffer stage n+1 is about to overwrite.
#if __has_builtin(__builtin_amdgcn_s_wait_asynccnt)
        __builtin_amdgcn_s_wait_asynccnt(0);
#else
        asm volatile("s_wait_asynccnt 0x0" ::: "memory");
#endif
#endif
        __syncthreads();
        if (n + 1 < nk) stage(buf ^ 1, (n + 1) * 32);

        // A fragments: lane holds row M=lrow, K runs {half*8..+7, 16+half*8..+7}
        v16bf af[4];
        #pragma unroll
        for (int mt = 0; mt < 4; mt++) {
            const unsigned short* p = &ldsA[buf][(wm + mt * 16 + lrow) * LDT];
            bf16x8 x0 = *(const bf16x8*)(p + half * 8);
            bf16x8 x1 = *(const bf16x8*)(p + 16 + half * 8);
            af[mt] = __builtin_shufflevector(x0, x1, 0,1,2,3,4,5,6,7,8,9,10,11,12,13,14,15);
        }
        // B fragments: lane holds col N=lrow, K runs {half*8..+7, 16+half*8..+7}
        v16bf bfr[2];
        #pragma unroll
        for (int nt = 0; nt < 2; nt++) {
            const unsigned short* p = &ldsW[buf][(wn + nt * 16 + lrow) * LDT];
            bf16x8 x0 = *(const bf16x8*)(p + half * 8);
            bf16x8 x1 = *(const bf16x8*)(p + 16 + half * 8);
            bfr[nt] = __builtin_shufflevector(x0, x1, 0,1,2,3,4,5,6,7,8,9,10,11,12,13,14,15);
        }
        #pragma unroll
        for (int mt = 0; mt < 4; mt++)
            #pragma unroll
            for (int nt = 0; nt < 2; nt++)
                acc[mt][nt] = __builtin_amdgcn_wmma_f32_16x16x32_bf16(
                    false, af[mt], false, bfr[nt], (short)0, acc[mt][nt], false, false);
    }

    // Epilogue: C/D layout -> lane writes col N=lrow, rows M = half*8 + v
    #pragma unroll
    for (int mt = 0; mt < 4; mt++) {
        #pragma unroll
        for (int nt = 0; nt < 2; nt++) {
            const long col  = blockN + wn + nt * 16 + lrow;
            const long rowb = blockM + wm + mt * 16 + half * 8;
            const float bv = HAS_BIAS ? bias[col] : 0.f;
            #pragma unroll
            for (int v = 0; v < 8; v++) {
                float val = acc[mt][nt][v] + bv;
                if (RELU) val = fmaxf(val, 0.f);
                const long idx = (rowb + v) * (long)N + col;
                if (ACCUM) out[idx] += val; else out[idx] = val;
            }
        }
    }
}

// ---------------------------------------------------------------------------
// climate = tanh(h @ w2^T + b2) : tiny (1024x64x128), plain f32
// ---------------------------------------------------------------------------
__global__ __launch_bounds__(64) void k_climate(
        const float* __restrict__ h, const float* __restrict__ w2,
        const float* __restrict__ b2, float* __restrict__ cl) {
    const int b = blockIdx.x, n = threadIdx.x;
    const float* hr = h  + (long)b * (2 * C_);
    const float* wr = w2 + (long)n * (2 * C_);
    float acc = b2[n];
    for (int k = 0; k < 2 * C_; k++) acc += hr[k] * wr[k];
    cl[(long)b * C_ + n] = tanhf(acc);
}

// ---------------------------------------------------------------------------
// xa[b,r] = T[b,RC+r] + sum_c T[b,r*C+c]*climate[b,c];
// u[b,r*C+c] = bf16(xa*climate), u[b,RC+r] = bf16(xa), pad to K2P with 0
// ---------------------------------------------------------------------------
__global__ __launch_bounds__(256) void k_xa_u(
        const float* __restrict__ T, const float* __restrict__ cl,
        unsigned short* __restrict__ u) {
    __shared__ float scl[C_];
    __shared__ float sxa[R_];
    const int b = blockIdx.x, t = threadIdx.x;
    if (t < C_) scl[t] = cl[(long)b * C_ + t];
    __syncthreads();
    if (t < R_) {
        const float* Tr = T + (long)b * NTP;
        float acc = Tr[RC + t];
        for (int c = 0; c < C_; c++) acc += Tr[t * C_ + c] * scl[c];
        sxa[t] = acc;
    }
    __syncthreads();
    unsigned short* ur = u + (long)b * K2P;
    for (int idx = t; idx < K2P; idx += 256) {
        float v;
        if (idx < RC)      v = sxa[idx >> 6] * scl[idx & 63];
        else if (idx < K2) v = sxa[idx - RC];
        else               v = 0.f;
        ur[idx] = f2bf(v);
    }
}

// ---------------------------------------------------------------------------
extern "C" void kernel_launch(void* const* d_in, const int* in_sizes, int n_in,
                              void* d_out, int out_size, void* d_ws, size_t ws_size,
                              hipStream_t stream) {
    const float* x    = (const float*)d_in[0];
    const float* bw   = (const float*)d_in[1];
    const float* w1   = (const float*)d_in[2];
    const float* b1   = (const float*)d_in[3];
    const float* w2   = (const float*)d_in[4];
    const float* b2   = (const float*)d_in[5];
    const float* wa   = (const float*)d_in[6];
    const float* ba   = (const float*)d_in[7];
    const float* wb   = (const float*)d_in[8];
    const float* bb   = (const float*)d_in[9];
    float* out = (float*)d_out;

    char* ws = (char*)d_ws;
    size_t ofs = 0;
    auto alloc = [&](size_t bytes) { char* p = ws + ofs; ofs = (ofs + bytes + 255) & ~(size_t)255; return p; };
    unsigned short* xbf   = (unsigned short*)alloc((size_t)B_ * I_ * 2);   // x as bf16
    unsigned short* xnbf  = (unsigned short*)alloc((size_t)B_ * I_ * 2);   // xn as bf16
    unsigned short* bwbf  = (unsigned short*)alloc((size_t)O_ * I_ * 2);   // base_weight bf16
    unsigned short* w1bf  = (unsigned short*)alloc((size_t)(2 * C_) * I_ * 2);
    unsigned short* wat   = (unsigned short*)alloc((size_t)NTP * I_ * 2);  // wa_t (padded)
    unsigned short* wbaug = (unsigned short*)alloc((size_t)O_ * K2P * 2);  // wb_aug (padded)
    float* h    = (float*)alloc((size_t)B_ * (2 * C_) * 4);
    float* clim = (float*)alloc((size_t)B_ * C_ * 4);
    float* Tbuf = (float*)alloc((size_t)B_ * NTP * 4);
    unsigned short* ubuf = (unsigned short*)alloc((size_t)B_ * K2P * 2);
    (void)ws_size; (void)in_sizes; (void)n_in; (void)out_size;

    // 1) standardize + bf16 conversions of x / xn
    k_standardize<<<B_, 256, 0, stream>>>(x, xbf, xnbf);
    // 2) weight conversions / reshapes
    k_f32_to_bf16<<<4096, 256, 0, stream>>>(bw, bwbf, (long)O_ * I_);
    k_f32_to_bf16<<<1024, 256, 0, stream>>>(w1, w1bf, (long)(2 * C_) * I_);
    k_build_wa_t<<<4096, 256, 0, stream>>>(wa, ba, wat);
    k_build_wb_aug<<<4096, 256, 0, stream>>>(wb, bb, wbaug);
    // 3) sensor layer 1: h = relu(xn @ w1^T + b1)   (M=1024, N=128, K=2048)
    k_gemm_bf16<false, true, true><<<dim3(1, 8), 256, 0, stream>>>(xnbf, w1bf, b1, h, 2 * C_, I_);
    // 4) climate = tanh(h @ w2^T + b2)
    k_climate<<<B_, 64, 0, stream>>>(h, w2, b2, clim);
    // 5) T = x @ wa_t^T   (M=1024, N=1152, K=2048)
    k_gemm_bf16<false, false, false><<<dim3(NTP / 128, 8), 256, 0, stream>>>(xbf, wat, nullptr, Tbuf, NTP, I_);
    // 6) xa + outer product -> u (bf16, padded K2P)
    k_xa_u<<<B_, 256, 0, stream>>>(Tbuf, clim, ubuf);
    // 7) base_out = x @ base_weight^T   (M=1024, N=2048, K=2048)
    k_gemm_bf16<false, false, false><<<dim3(O_ / 128, 8), 256, 0, stream>>>(xbf, bwbf, nullptr, out, O_, I_);
    // 8) out += u @ wb_aug^T   (M=1024, N=2048, K=1056)
    k_gemm_bf16<true, false, false><<<dim3(O_ / 128, 8), 256, 0, stream>>>(ubuf, wbaug, nullptr, out, O_, K2P);
}
